// GAT_4621384810581
// MI455X (gfx1250) — compile-verified, matrix operand order checked by reading, not run on hardware
//
#include <hip/hip_runtime.h>

// ---------------------------------------------------------------------------
// GAT (2-layer, 8 heads) for MI455X / gfx1250.
// Dense GEMMs: v_wmma_f32_16x16x32_bf16 (wave32 WMMA, fp32 accum).
// Tile staging: async global->LDS copies (global_load_async_to_lds_b64,
// ASYNCcnt) with double-buffered LDS; register-pipelined synchronous
// fallback. Sparse attention: float atomic-max bit trick + atomicAdd.
// ---------------------------------------------------------------------------

typedef __attribute__((ext_vector_type(16))) __bf16 v16bf;
typedef __attribute__((ext_vector_type(8)))  float  v8f;

#define GAT_N      50000
#define GAT_E      800000
#define GAT_NFEAT  512
#define GAT_NHID   64
#define GAT_NHEADS 8
#define GAT_NCLASS 64
#define GAT_ALPHA  0.2f

#if defined(__has_builtin)
# if __has_builtin(__builtin_amdgcn_global_load_async_to_lds_b64)
#  define GAT_HAVE_ASYNC_CP 1
# endif
#endif
#ifndef GAT_HAVE_ASYNC_CP
# define GAT_HAVE_ASYNC_CP 0
#endif

#if GAT_HAVE_ASYNC_CP
// Builtin signature (probe-confirmed): (int2 as(1)* gsrc, int2 as(3)* ldst,
// imm offset, cpol). as(3) pointer value == low 32 bits of the generic LDS
// pointer (LDS aperture lives in addr[63:32] on CDNA5).
typedef int gat_v2i __attribute__((vector_size(8)));
typedef gat_v2i __attribute__((address_space(1)))* gat_gp;
typedef gat_v2i __attribute__((address_space(3)))* gat_lp;

__device__ __forceinline__ void gat_async_cp8(const void* g, void* l) {
  __builtin_amdgcn_global_load_async_to_lds_b64(
      (gat_gp)(unsigned long long)g,
      (gat_lp)(unsigned int)(unsigned long long)l,
      0, 0);
}
__device__ __forceinline__ void gat_wait_async() {
# if __has_builtin(__builtin_amdgcn_s_wait_asynccnt)
  __builtin_amdgcn_s_wait_asynccnt(0);
# else
  asm volatile("s_wait_asynccnt 0x0" ::: "memory");
# endif
}
#endif

static __device__ __forceinline__ unsigned short gat_bf16_bits(float f) {
  __bf16 b = (__bf16)f;
  union { __bf16 h; unsigned short u; } cv;
  cv.h = b;
  return cv.u;
}

// ---------------------------------------------------------------------------
// Utility kernels
// ---------------------------------------------------------------------------
__global__ void gat_fill_f32(float* __restrict__ p, float v, int n) {
  for (int i = blockIdx.x * blockDim.x + threadIdx.x; i < n;
       i += gridDim.x * blockDim.x)
    p[i] = v;
}

// f32 -> bf16, 4 elements per thread (n must be a multiple of 4)
__global__ void gat_cvt_bf16x4(const float4* __restrict__ in,
                               ushort4* __restrict__ out, int n4) {
  for (int i = blockIdx.x * blockDim.x + threadIdx.x; i < n4;
       i += gridDim.x * blockDim.x) {
    float4 f = in[i];
    ushort4 o;
    o.x = gat_bf16_bits(f.x);
    o.y = gat_bf16_bits(f.y);
    o.z = gat_bf16_bits(f.z);
    o.w = gat_bf16_bits(f.w);
    out[i] = o;
  }
}

// ELU + convert: h1b = bf16(elu(h1))   (h1 stays fp32 accumulator)
__global__ void gat_elu_cvt(const float* __restrict__ in,
                            unsigned short* __restrict__ out, int n) {
  for (int i = blockIdx.x * blockDim.x + threadIdx.x; i < n;
       i += gridDim.x * blockDim.x) {
    float v = in[i];
    v = v > 0.0f ? v : (__expf(v) - 1.0f);
    out[i] = gat_bf16_bits(v);
  }
}

// Pack W [8,512,64] -> bf16 Bt1 [col c = h*64+j][k] = [512][512]
__global__ void gat_pack_W1t(const float* __restrict__ W,
                             unsigned short* __restrict__ Bt1) {
  int i = blockIdx.x * blockDim.x + threadIdx.x;
  if (i >= 512 * 512) return;
  int c = i >> 9;
  int k = i & 511;
  int h = c >> 6;
  int j = c & 63;
  Bt1[i] = gat_bf16_bits(W[h * (512 * 64) + k * 64 + j]);
}

// Pack W_out [512,64] -> bf16 Bt2 [col j][k] = [64][512]
__global__ void gat_pack_W2t(const float* __restrict__ Wo,
                             unsigned short* __restrict__ Bt2) {
  int i = blockIdx.x * blockDim.x + threadIdx.x;
  if (i >= 64 * 512) return;
  int j = i >> 9;
  int k = i & 511;
  Bt2[i] = gat_bf16_bits(Wo[k * 64 + j]);
}

// ---------------------------------------------------------------------------
// Tiled GEMM: C[M x Nc] (f32) = A[M x K] (bf16) * B (bf16, stored [Nc][K])
// Block tile 128 x (NT*16), 8 waves, wave tile 16 x (NT*16), K step 32.
// Fragment layout per CDNA5 ISA 7.12.2 (16-bit 16x32 A / 32x16 B):
//   lanes 0-15 : M(or N)=lane,    halves 0-7 = K 0-7,  8-15 = K 16-23
//   lanes 16-31: M(or N)=lane-16, halves 0-7 = K 8-15, 8-15 = K 24-31
// Both A and B tiles are [row-or-col][k] in LDS -> fragments = 2x ds_load_b128.
// All NT B-fragments are preloaded into distinct registers so the ds_load
// clause overlaps with the WMMA chain (partial dscnt waits).
// ---------------------------------------------------------------------------
template <int NT>   // number of 16-wide N tiles per block (4 or 8)
__global__ __launch_bounds__(256)
void gat_gemm_bf16_wmma(const unsigned short* __restrict__ Ab,
                        const unsigned short* __restrict__ Bt,
                        float* __restrict__ C, int M, int K, int Nc) {
  const int tid  = threadIdx.x;
  const int lane = tid & 31;
  const int w    = tid >> 5;                      // wave id 0..7
  const int rowBase = blockIdx.x * 128;
  const int colBase = blockIdx.y * (NT * 16);

  v8f acc[NT] = {};

  const int mLocal = w * 16 + (lane & 15);        // A-fragment row in tile
  const int kb0    = (lane < 16) ? 0 : 8;         // per-lane K phase

  union Frag { uint4 q[2]; v16bf v; };

#if GAT_HAVE_ASYNC_CP
  // ---------------- async copy path: double-buffered LDS -----------------
  __shared__ __align__(16) unsigned short lA[2][128 * 32];
  __shared__ __align__(16) unsigned short lB[2][NT * 16 * 32];

  auto stage = [&](int kb, int buf) {
#pragma unroll
    for (int i = 0; i < 4; ++i) {                 // A: 128x32 = 1024 quads
      int q = tid + i * 256;
      int r = q >> 3, kq = (q & 7) * 4;
      int gr = rowBase + r;
      gr = gr < M ? gr : (M - 1);                 // clamp; epilogue guards
      gat_async_cp8(&Ab[(long long)gr * K + kb + kq], &lA[buf][r * 32 + kq]);
    }
#pragma unroll
    for (int i = 0; i < NT / 2; ++i) {            // B: NT*128 quads
      int q = tid + i * 256;
      int c = q >> 3, kq = (q & 7) * 4;
      gat_async_cp8(&Bt[(long long)(colBase + c) * K + kb + kq],
                    &lB[buf][c * 32 + kq]);
    }
  };

  stage(0, 0);
  gat_wait_async();
  __syncthreads();

  for (int kb = 0; kb < K; kb += 32) {
    const int cur = (kb >> 5) & 1;
    if (kb + 32 < K) stage(kb + 32, cur ^ 1);     // overlap with compute

    Frag fa;
    fa.q[0] = *(const uint4*)&lA[cur][mLocal * 32 + kb0];
    fa.q[1] = *(const uint4*)&lA[cur][mLocal * 32 + kb0 + 16];

    Frag fb[NT];
#pragma unroll
    for (int nt = 0; nt < NT; ++nt) {             // issue all ds loads first
      int nLocal = nt * 16 + (lane & 15);
      fb[nt].q[0] = *(const uint4*)&lB[cur][nLocal * 32 + kb0];
      fb[nt].q[1] = *(const uint4*)&lB[cur][nLocal * 32 + kb0 + 16];
    }
#pragma unroll
    for (int nt = 0; nt < NT; ++nt)
      acc[nt] = __builtin_amdgcn_wmma_f32_16x16x32_bf16(
          false, fa.v, false, fb[nt].v, (short)0, acc[nt], false, false);

    gat_wait_async();                             // next buffer filled
    __syncthreads();                              // everyone done with cur
  }
#else
  // ------------- fallback: register-pipelined synchronous path -----------
  __shared__ __align__(16) unsigned short lA[128 * 32];
  __shared__ __align__(16) unsigned short lB[NT * 16 * 32];

  ushort4 ra[4], rb[NT / 2], ra2[4], rb2[NT / 2];

  auto loadA = [&](int kb, ushort4* dst) {
#pragma unroll
    for (int i = 0; i < 4; ++i) {
      int q = tid + i * 256;
      int r = q >> 3, kq = (q & 7) * 4;
      int gr = rowBase + r;
      gr = gr < M ? gr : (M - 1);
      dst[i] = *(const ushort4*)&Ab[(long long)gr * K + kb + kq];
    }
  };
  auto loadB = [&](int kb, ushort4* dst) {
#pragma unroll
    for (int i = 0; i < NT / 2; ++i) {
      int q = tid + i * 256;
      int c = q >> 3, kq = (q & 7) * 4;
      dst[i] = *(const ushort4*)&Bt[(long long)(colBase + c) * K + kb + kq];
    }
  };

  loadA(0, ra);
  loadB(0, rb);

  for (int kb = 0; kb < K; kb += 32) {
    // commit current tile to LDS (waits only here)
#pragma unroll
    for (int i = 0; i < 4; ++i) {
      int q = tid + i * 256;
      int r = q >> 3, kq = (q & 7) * 4;
      *(ushort4*)&lA[r * 32 + kq] = ra[i];
    }
#pragma unroll
    for (int i = 0; i < NT / 2; ++i) {
      int q = tid + i * 256;
      int c = q >> 3, kq = (q & 7) * 4;
      *(ushort4*)&lB[c * 32 + kq] = rb[i];
    }
    __syncthreads();

    // issue next tile's loads; wait deferred to next iteration's LDS stores
    if (kb + 32 < K) {
      loadA(kb + 32, ra2);
      loadB(kb + 32, rb2);
    }

    Frag fa;
    fa.q[0] = *(const uint4*)&lA[mLocal * 32 + kb0];
    fa.q[1] = *(const uint4*)&lA[mLocal * 32 + kb0 + 16];

    Frag fb[NT];
#pragma unroll
    for (int nt = 0; nt < NT; ++nt) {
      int nLocal = nt * 16 + (lane & 15);
      fb[nt].q[0] = *(const uint4*)&lB[nLocal * 32 + kb0];
      fb[nt].q[1] = *(const uint4*)&lB[nLocal * 32 + kb0 + 16];
    }
#pragma unroll
    for (int nt = 0; nt < NT; ++nt)
      acc[nt] = __builtin_amdgcn_wmma_f32_16x16x32_bf16(
          false, fa.v, false, fb[nt].v, (short)0, acc[nt], false, false);
    __syncthreads();

#pragma unroll
    for (int i = 0; i < 4; ++i) ra[i] = ra2[i];
#pragma unroll
    for (int i = 0; i < NT / 2; ++i) rb[i] = rb2[i];
  }
#endif

  // ---- store: C/D layout -> lanes 0-15: M=r, lanes 16-31: M=r+8 ----
  const int mAdd = (lane < 16) ? 0 : 8;
#pragma unroll
  for (int nt = 0; nt < NT; ++nt) {
    int gn = colBase + nt * 16 + (lane & 15);
#pragma unroll
    for (int r = 0; r < 8; ++r) {
      int gm = rowBase + w * 16 + r + mAdd;
      if (gm < M) C[(long long)gm * Nc + gn] = acc[nt][r];
    }
  }
}

// ---------------------------------------------------------------------------
// Per-node attention score halves: fd[n,h] = Wh[n,h*F:...]·a[h,0:F],
//                                  fs[n,h] = Wh[n,h*F:...]·a[h,F:2F]
// ---------------------------------------------------------------------------
__global__ void gat_scores(const float* __restrict__ Wh, const float* __restrict__ a,
                           float* __restrict__ fd, float* __restrict__ fs,
                           int Nn, int H, int F, int stride) {
  int i = blockIdx.x * blockDim.x + threadIdx.x;
  if (i >= Nn * H) return;
  int n = i / H, h = i - n * H;
  const float* wp = Wh + (long long)n * stride + h * F;
  const float* ad = a + h * 2 * F;
  const float* as = ad + F;
  float sd = 0.0f, ss = 0.0f;
  for (int j = 0; j < F; ++j) {
    float wv = wp[j];
    sd += wv * ad[j];
    ss += wv * as[j];
  }
  fd[i] = sd;
  fs[i] = ss;
}

// ---------------------------------------------------------------------------
// Float atomic-max via monotone int/uint bit trick (init -inf safe):
//   v >= 0: signed-int max on raw bits; v < 0: unsigned-int min on raw bits.
// ---------------------------------------------------------------------------
__device__ __forceinline__ void gat_atomicMaxF(float* addr, float v) {
  if (v >= 0.0f) atomicMax((int*)addr, __float_as_int(v));
  else           atomicMin((unsigned int*)addr, __float_as_uint(v));
}

__device__ __forceinline__ float gat_lrelu(float x) {
  return x > 0.0f ? x : GAT_ALPHA * x;
}

__global__ void gat_edge_max(const int* __restrict__ row, const int* __restrict__ col,
                             const float* __restrict__ fd, const float* __restrict__ fs,
                             float* __restrict__ m, int E, int H) {
  int total = E * H;
  for (int i = blockIdx.x * blockDim.x + threadIdx.x; i < total;
       i += gridDim.x * blockDim.x) {
    int e = i / H, h = i - e * H;
    int r = row[e], c = col[e];
    float ev = gat_lrelu(fd[r * H + h] + fs[c * H + h]);
    gat_atomicMaxF(&m[r * H + h], ev);
  }
}

__global__ void gat_edge_sum(const int* __restrict__ row, const int* __restrict__ col,
                             const float* __restrict__ fd, const float* __restrict__ fs,
                             const float* __restrict__ m, float* __restrict__ den,
                             int E, int H) {
  int total = E * H;
  for (int i = blockIdx.x * blockDim.x + threadIdx.x; i < total;
       i += gridDim.x * blockDim.x) {
    int e = i / H, h = i - e * H;
    int r = row[e], c = col[e];
    float ev = gat_lrelu(fd[r * H + h] + fs[c * H + h]);
    float p = __expf(ev - m[r * H + h]);
    atomicAdd(&den[r * H + h], p);
  }
}

// One wave per (edge, head); lanes cover the F=64 features (2 each).
__global__ __launch_bounds__(256)
void gat_edge_aggregate(const int* __restrict__ row, const int* __restrict__ col,
                        const float* __restrict__ fd, const float* __restrict__ fs,
                        const float* __restrict__ m, const float* __restrict__ den,
                        const float* __restrict__ Wh, float* __restrict__ out,
                        int E, int H, int F, int stride) {
  int lane = threadIdx.x & 31;
  int wid  = (blockIdx.x * blockDim.x + threadIdx.x) >> 5;
  int nw   = (gridDim.x * blockDim.x) >> 5;
  int total = E * H;
  for (int wrk = wid; wrk < total; wrk += nw) {
    int e = wrk / H, h = wrk - e * H;
    int r = row[e], c = col[e];
    float ev  = gat_lrelu(fd[r * H + h] + fs[c * H + h]);
    float p   = __expf(ev - m[r * H + h]);
    float att = p / fmaxf(den[r * H + h], 1e-16f);
    const float* src = Wh + (long long)c * stride + h * F;
    float*       dst = out + (long long)r * stride + h * F;
    for (int j = lane; j < F; j += 32)
      atomicAdd(&dst[j], att * src[j]);
  }
}

// ---------------------------------------------------------------------------
// Launch
// ---------------------------------------------------------------------------
extern "C" void kernel_launch(void* const* d_in, const int* in_sizes, int n_in,
                              void* d_out, int out_size, void* d_ws, size_t ws_size,
                              hipStream_t stream) {
  const float* x     = (const float*)d_in[0];   // [N, 512]
  const int*   row   = (const int*)d_in[1];     // [E]
  const int*   col   = (const int*)d_in[2];     // [E]
  const float* W     = (const float*)d_in[3];   // [8, 512, 64]
  const float* a     = (const float*)d_in[4];   // [8, 128]
  const float* W_out = (const float*)d_in[5];   // [512, 64]
  const float* a_out = (const float*)d_in[6];   // [128]
  float* out = (float*)d_out;                   // [N, 64]

  const int N = GAT_N, E = GAT_E, H = GAT_NHEADS;
  const int F1 = GAT_NFEAT;                 // 512
  const int FC = GAT_NHEADS * GAT_NHID;     // 512 (concat width)
  const int F2 = GAT_NCLASS;                // 64

  // ---- workspace layout ----
  char*  ws  = (char*)d_ws;
  size_t off = 0;
  auto alloc = [&](size_t bytes) -> void* {
    void* p = ws + off;
    off += (bytes + 255) & ~(size_t)255;
    return p;
  };
  unsigned short* Ab   = (unsigned short*)alloc((size_t)N * F1 * 2);  // bf16(x)
  unsigned short* Bt1  = (unsigned short*)alloc((size_t)FC * F1 * 2); // W packed [col][k]
  unsigned short* Bt2  = (unsigned short*)alloc((size_t)F2 * FC * 2); // W_out packed
  unsigned short* h1b  = (unsigned short*)alloc((size_t)N * FC * 2);  // bf16(elu(h1))
  float*  Wh1  = (float*) alloc((size_t)N * FC * 4);                  // x @ W (all heads)
  float*  h1   = (float*) alloc((size_t)N * FC * 4);                  // layer-1 accum
  float*  Wh2  = (float*) alloc((size_t)N * F2 * 4);                  // h1 @ W_out
  float*  fd1  = (float*) alloc((size_t)N * H * 4);
  float*  fs1  = (float*) alloc((size_t)N * H * 4);
  float*  m1   = (float*) alloc((size_t)N * H * 4);
  float*  den1 = (float*) alloc((size_t)N * H * 4);
  float*  fd2  = (float*) alloc((size_t)N * 4);
  float*  fs2  = (float*) alloc((size_t)N * 4);
  float*  m2   = (float*) alloc((size_t)N * 4);
  float*  den2 = (float*) alloc((size_t)N * 4);

  const float NEG_INF = -__builtin_inff();

  // ---- convert / pack operands to bf16 ----
  gat_cvt_bf16x4<<<4096, 256, 0, stream>>>((const float4*)x, (ushort4*)Ab,
                                           N * F1 / 4);
  gat_pack_W1t<<<(FC * F1 + 255) / 256, 256, 0, stream>>>(W, Bt1);
  gat_pack_W2t<<<(F2 * FC + 255) / 256, 256, 0, stream>>>(W_out, Bt2);

  // ==================== Layer 1 ====================
  // Wh1 = x @ W_all  (M=50000, K=512, Nc=512), block tile 128x128
  gat_gemm_bf16_wmma<8><<<dim3((N + 127) / 128, FC / 128), 256, 0, stream>>>(
      Ab, Bt1, Wh1, N, F1, FC);

  gat_scores<<<(N * H + 255) / 256, 256, 0, stream>>>(
      Wh1, a, fd1, fs1, N, H, GAT_NHID, FC);

  gat_fill_f32<<<1024, 256, 0, stream>>>(m1, NEG_INF, N * H);
  gat_fill_f32<<<1024, 256, 0, stream>>>(den1, 0.0f, N * H);
  gat_fill_f32<<<4096, 256, 0, stream>>>(h1, 0.0f, N * FC);

  gat_edge_max<<<2048, 256, 0, stream>>>(row, col, fd1, fs1, m1, E, H);
  gat_edge_sum<<<2048, 256, 0, stream>>>(row, col, fd1, fs1, m1, den1, E, H);
  gat_edge_aggregate<<<4096, 256, 0, stream>>>(
      row, col, fd1, fs1, m1, den1, Wh1, h1, E, H, GAT_NHID, FC);

  // ELU + bf16 conversion for the layer-2 GEMM operand
  gat_elu_cvt<<<4096, 256, 0, stream>>>(h1, h1b, N * FC);

  // ==================== Layer 2 ====================
  // Wh2 = h1 @ W_out  (M=50000, K=512, Nc=64), block tile 128x64
  gat_gemm_bf16_wmma<4><<<dim3((N + 127) / 128, F2 / 64), 256, 0, stream>>>(
      h1b, Bt2, Wh2, N, FC, F2);

  gat_scores<<<(N + 255) / 256, 256, 0, stream>>>(
      Wh2, a_out, fd2, fs2, N, 1, F2, F2);

  gat_fill_f32<<<1024, 256, 0, stream>>>(m2, NEG_INF, N);
  gat_fill_f32<<<1024, 256, 0, stream>>>(den2, 0.0f, N);
  gat_fill_f32<<<2048, 256, 0, stream>>>(out, 0.0f, N * F2);

  gat_edge_max<<<2048, 256, 0, stream>>>(row, col, fd2, fs2, m2, E, 1);
  gat_edge_sum<<<2048, 256, 0, stream>>>(row, col, fd2, fs2, m2, den2, E, 1);
  gat_edge_aggregate<<<4096, 256, 0, stream>>>(
      row, col, fd2, fs2, m2, den2, Wh2, out, E, 1, F2, F2);
  // concat=False: no final activation.
}
